// ShiftPriorityTopKGate_53523882443556
// MI455X (gfx1250) — matrix operand order repaired and checked by reading, not compiled, not run on hardware
//
#include <hip/hip_runtime.h>

#define S 8192
#define D 1024
#define E 64
#define CAP 128
#define NBIG 67108864LL   // S*E*CAP

typedef __attribute__((ext_vector_type(16))) _Float16 v16h;
typedef __attribute__((ext_vector_type(8)))  float    v8f;

// ---------------- K0: zero small ws accumulators ----------------
__global__ void init_ws_kernel(float* me) {
    if (threadIdx.x < E) me[threadIdx.x] = 0.0f;
}

// Build one 16x16x32-f16 A/B fragment pair for K-chunk starting at k0.
// Layouts per ISA 7.12.2 (wave32):
//  A (16x32): lane-lo holds K 0..7 & 16..23, lane-hi K 8..15 & 24..31, M = lane&15
//  B (32x16): lane-lo holds K 0..7 & 16..23, lane-hi K 8..15 & 24..31, N = lane&15
__device__ __forceinline__ void load_frags(const float* __restrict__ xrow,
                                           const float* __restrict__ wrow,
                                           int k0, int hi, v16h& a, v16h& b) {
    const float* pa0 = xrow + k0 + hi * 8;    // 8 contiguous f32
    const float* pa1 = pa0 + 16;              // 8 contiguous f32
    const float* pb  = wrow + k0 + hi * 8;    // 8 + 8 f32 (second at +16)
    #pragma unroll
    for (int e = 0; e < 8; ++e) {
        a[e]     = (_Float16)pa0[e];
        a[e + 8] = (_Float16)pa1[e];
    }
    #pragma unroll
    for (int e = 0; e < 16; ++e) b[e] = (_Float16)pb[e < 8 ? e : e + 8];
}

// ---------------- K1: logits = x @ W^T via WMMA f16 ----------------
// grid = S/16 blocks, block = 128 threads (4 waves).
// wave w computes the 16x16 tile (rows row0..row0+15, experts 16w..16w+15).
// K is reduced in 64-wide steps into TWO independent accumulators so the
// v_wmma chain has no serial D->C RAW each iteration (hazard table 7.12.1).
__global__ __launch_bounds__(128) void gemm_logits_kernel(
        const float* __restrict__ x, const float* __restrict__ w,
        float* __restrict__ logits) {
    const int wave = threadIdx.x >> 5;
    const int lane = threadIdx.x & 31;
    const int hi   = lane >> 4;
    const int l15  = lane & 15;
    const int row0 = blockIdx.x * 16;
    const int col0 = wave * 16;

    const float* __restrict__ xrow = x + (size_t)(row0 + l15) * D;
    const float* __restrict__ wrow = w + (size_t)(col0 + l15) * D;

    v8f c0 = {}, c1 = {};
    for (int k0 = 0; k0 < D; k0 += 64) {
        if (k0 + 64 < D) {
            __builtin_prefetch(xrow + k0 + 64, 0, 1);
            __builtin_prefetch(wrow + k0 + 64, 0, 1);
        }
        v16h a0, b0, a1, b1;
        load_frags(xrow, wrow, k0,      hi, a0, b0);
        load_frags(xrow, wrow, k0 + 32, hi, a1, b1);
        c0 = __builtin_amdgcn_wmma_f32_16x16x32_f16(
                false, a0, false, b0, (short)0, c0, false, false);
        c1 = __builtin_amdgcn_wmma_f32_16x16x32_f16(
                false, a1, false, b1, (short)0, c1, false, false);
    }

    // C/D layout: VGPR r -> M = r + 8*hi, N = lane&15.
    #pragma unroll
    for (int r = 0; r < 8; ++r) {
        int row = row0 + r + hi * 8;
        logits[(size_t)row * E + col0 + l15] = c0[r] + c1[r];
    }
}

// ---------------- K2: softmax + argmax + per-expert gate sums ----------------
// block = 256 threads = 8 waves, one token per wave, 2 experts per lane.
__global__ __launch_bounds__(256) void softmax_stats_kernel(
        const float* __restrict__ logits, float* __restrict__ gates,
        float* __restrict__ mx, int* __restrict__ amax, float* __restrict__ me) {
    __shared__ float sme[E];
    const int tid = threadIdx.x;
    if (tid < E) sme[tid] = 0.0f;
    __syncthreads();

    const int wave = tid >> 5;
    const int lane = tid & 31;
    const int t = blockIdx.x * 8 + wave;
    const float* lrow = logits + (size_t)t * E;

    float v0 = lrow[lane];
    float v1 = lrow[lane + 32];
    float bv; int bi;
    if (v0 >= v1) { bv = v0; bi = lane; } else { bv = v1; bi = lane + 32; }
    #pragma unroll
    for (int off = 16; off; off >>= 1) {
        float ov = __shfl_xor(bv, off, 32);
        int   oi = __shfl_xor(bi, off, 32);
        if (ov > bv || (ov == bv && oi < bi)) { bv = ov; bi = oi; }
    }
    float e0 = __expf(v0 - bv), e1 = __expf(v1 - bv);
    float s = e0 + e1;
    #pragma unroll
    for (int off = 16; off; off >>= 1) s += __shfl_xor(s, off, 32);
    const float inv = 1.0f / s;          // == max gate value
    const float g0 = e0 * inv, g1 = e1 * inv;

    gates[(size_t)t * E + lane]      = g0;
    gates[(size_t)t * E + lane + 32] = g1;
    atomicAdd(&sme[lane], g0);
    atomicAdd(&sme[lane + 32], g1);
    if (lane == 0) { mx[t] = inv; amax[t] = bi; }

    __syncthreads();
    if (tid < E) atomicAdd(&me[tid], sme[tid]);
}

// ---------------- K3: stable rank (desc max-gate) + roll + order scatter ----
__global__ __launch_bounds__(256) void rank_pos_kernel(
        const float* __restrict__ mx, int* __restrict__ pos,
        int* __restrict__ order) {
    __shared__ float tile[256];
    const int t = blockIdx.x * 256 + threadIdx.x;
    const float key = mx[t];
    int rank = 0;
    for (int base = 0; base < S; base += 256) {
        tile[threadIdx.x] = mx[base + threadIdx.x];
        __syncthreads();
        #pragma unroll 8
        for (int j = 0; j < 256; ++j) {
            float o = tile[j];
            int idx = base + j;
            rank += (o > key) || ((o == key) && (idx < t));
        }
        __syncthreads();
    }
    const int p = (rank + S / 2) & (S - 1);   // jnp.roll by S*SHIFT_PRIORITY
    pos[t] = p;
    order[p] = t;
}

// ---------------- K4: per-expert positional slot assignment ----------------
// single block of 64 threads; thread e walks the rolled order for expert e.
__global__ void assign_locs_kernel(const int* __restrict__ order,
        const int* __restrict__ amax, int* __restrict__ loc,
        int* __restrict__ expcnt) {
    const int e = threadIdx.x;
    int cnt = 0;
    for (int i = 0; i < S; ++i) {
        int t = order[i];
        if (amax[t] == e) { loc[t] = cnt; ++cnt; }
    }
    expcnt[e] = cnt;   // pre-drop count (== exp_counts)
}

// ---------------- K5: float4 zero-fill of the whole output ----------------
__global__ __launch_bounds__(256) void fill_zero_kernel(float* __restrict__ out,
        long long n) {
    const long long stride = (long long)gridDim.x * blockDim.x;
    long long i = (long long)blockIdx.x * blockDim.x + threadIdx.x;
    const long long n4 = n >> 2;
    float4* p = (float4*)out;
    const float4 z = make_float4(0.f, 0.f, 0.f, 0.f);
    for (long long q = i; q < n4; q += stride) p[q] = z;
    if (i == 0)
        for (long long r = n4 << 2; r < n; ++r) out[r] = 0.0f;
}

// ---------------- K6: scalars (l_aux, exp_counts, unrouted rate) -----------
__global__ void finalize_kernel(const float* __restrict__ me,
        const int* __restrict__ expcnt, float* __restrict__ out) {
    __shared__ float saux[E];
    __shared__ int sdrop[E];
    const int e = threadIdx.x;
    const float mef = me[e] / (float)S;
    const float cef = (float)expcnt[e] / (float)S;
    saux[e] = mef * cef;
    int d = expcnt[e] - CAP;
    sdrop[e] = d > 0 ? d : 0;
    out[1 + 2 * NBIG + e] = (float)expcnt[e];
    __syncthreads();
    #pragma unroll
    for (int off = 32; off; off >>= 1) {
        if (e < off) { saux[e] += saux[e + off]; sdrop[e] += sdrop[e + off]; }
        __syncthreads();
    }
    if (e == 0) {
        out[0] = saux[0] * (float)E * 0.01f;                 // l_aux
        out[1 + 2 * NBIG + E] = (float)sdrop[0] / (float)S;  // unrouted rate
    }
}

// ---------------- K7: scatter the <=8192 nonzeros ----------------
__global__ __launch_bounds__(256) void scatter_kernel(
        const float* __restrict__ gates, const int* __restrict__ amax,
        const int* __restrict__ loc, float* __restrict__ out) {
    const int t = blockIdx.x * 256 + threadIdx.x;
    const int e = amax[t];
    const int l = loc[t];
    if (l < CAP) {
        float g = gates[(size_t)t * E + e];
        long long off = 1LL + ((long long)t * E + e) * CAP + l;
        out[off] = g;              // combine_weights[t, e, l]
        out[off + NBIG] = 1.0f;    // dispatch_mask[t, e, l]
    }
}

extern "C" void kernel_launch(void* const* d_in, const int* in_sizes, int n_in,
                              void* d_out, int out_size, void* d_ws, size_t ws_size,
                              hipStream_t stream) {
    const float* x = (const float*)d_in[0];      // [S, D]
    const float* w = (const float*)d_in[1];      // [E, D]
    float* out = (float*)d_out;

    // workspace layout
    float* gates  = (float*)d_ws;                 // S*E (holds logits then gates)
    float* mx     = gates + (size_t)S * E;        // S
    int*   amax   = (int*)(mx + S);               // S
    int*   pos    = amax + S;                     // S
    int*   order  = pos + S;                      // S
    int*   loc    = order + S;                    // S
    float* me     = (float*)(loc + S);            // E
    int*   expcnt = (int*)(me + E);               // E

    init_ws_kernel<<<1, 64, 0, stream>>>(me);
    gemm_logits_kernel<<<S / 16, 128, 0, stream>>>(x, w, gates);
    softmax_stats_kernel<<<S / 8, 256, 0, stream>>>(gates, gates, mx, amax, me);
    rank_pos_kernel<<<S / 256, 256, 0, stream>>>(mx, pos, order);
    assign_locs_kernel<<<1, 64, 0, stream>>>(order, amax, loc, expcnt);
    fill_zero_kernel<<<8192, 256, 0, stream>>>(out, (long long)out_size);
    finalize_kernel<<<1, 64, 0, stream>>>(me, expcnt, out);
    scatter_kernel<<<S / 256, 256, 0, stream>>>(gates, amax, loc, out);
}
